// SparseAttention_47278999994875
// MI455X (gfx1250) — matrix operand order repaired and checked by reading, not compiled, not run on hardware
//
#include <hip/hip_runtime.h>

// ---------------------------------------------------------------------------
// Problem constants (match reference)
// ---------------------------------------------------------------------------
#define DIMX   2048
#define HD     128
#define NH     16
#define NKV    4
#define BSZ    2
#define SEQ    2048
#define M_TOT  (BSZ * SEQ)     // 4096 rows (b,s)
#define KVD    (NKV * HD)      // 512

typedef __attribute__((ext_vector_type(16))) __bf16 v16bf;
typedef __attribute__((ext_vector_type(8)))  float  v8f;
typedef __attribute__((ext_vector_type(4)))  int    i32x4v;

#define WMMA_BF16(A, B, C) \
    __builtin_amdgcn_wmma_f32_16x16x32_bf16(false, (A), false, (B), (short)0, (C), false, false)

__device__ __forceinline__ v16bf ldfrag(const __bf16* p) {
    return *(const v16bf*)p;   // 32B per lane -> b128-class loads
}

// ---------------------------------------------------------------------------
// Async global->LDS path (CDNA5), guarded so compilation never breaks.
// ---------------------------------------------------------------------------
#if defined(__has_builtin)
#  if __has_builtin(__builtin_amdgcn_global_load_async_to_lds_b128) && \
      __has_builtin(__builtin_amdgcn_s_wait_asynccnt)
#    define USE_ASYNC_LDS 1
#  endif
#endif

typedef __attribute__((address_space(1))) i32x4v* gb128_t;   // global 16B vec ptr
typedef __attribute__((address_space(3))) i32x4v* lb128_t;   // LDS 16B vec ptr

__device__ __forceinline__ void copy16_g2l(const __bf16* src, __bf16* dst) {
#ifdef USE_ASYNC_LDS
    __builtin_amdgcn_global_load_async_to_lds_b128(
        (gb128_t)(void*)const_cast<__bf16*>(src),
        (lb128_t)(void*)dst, 0, 0);
#else
    *(int4*)dst = *(const int4*)src;
#endif
}

__device__ __forceinline__ void wait_async_lds() {
#ifdef USE_ASYNC_LDS
    __builtin_amdgcn_s_wait_asynccnt(0);
#endif
}

// ---------------------------------------------------------------------------
// f32 -> bf16 conversion (vectorized)
// ---------------------------------------------------------------------------
__global__ void cvt_bf16_kernel(const float* __restrict__ s, __bf16* __restrict__ d, int n4) {
    int i = blockIdx.x * blockDim.x + threadIdx.x;
    int stride = gridDim.x * blockDim.x;
    for (; i < n4; i += stride) {
        float4 f = ((const float4*)s)[i];
        __bf16* o = d + (size_t)i * 4;
        o[0] = (__bf16)f.x; o[1] = (__bf16)f.y;
        o[2] = (__bf16)f.z; o[3] = (__bf16)f.w;
    }
}

// ---------------------------------------------------------------------------
// C[M,N] = A[M,K] * B[N,K]^T   (A,B bf16 row-major, C f32)
// Block = 8 waves, tile M=16 x N=512.  A tile staged through LDS in
// double-buffered 128-col stages (async to LDS when available); each wave
// computes a 16x64 strip with register-pipelined B fragments.
// grid = (N/512, M/16), block = 256
// ---------------------------------------------------------------------------
#define STAGE_COLS 128
#define LDS_ROW    136                 // 128 + 8 bf16 pad (16B) -> bank-shift
__global__ void __launch_bounds__(256)
gemm_bt_bf16(const __bf16* __restrict__ A, const __bf16* __restrict__ Bm,
             float* __restrict__ C, int M, int N, int K) {
    __shared__ __align__(16) __bf16 atile[2][16 * LDS_ROW];

    const int tid  = threadIdx.x;
    const int lane = tid & 31;
    const int wave = tid >> 5;
    const int m0   = blockIdx.y * 16;
    const int n0   = (blockIdx.x * 8 + wave) * 64;

    const int row  = lane & 15;
    const int half = lane >> 4;

    // stage loader: 256 threads x 16B = 4KB = 16 rows x 128 cols bf16
    const int lrow = tid >> 4;          // 0..15
    const int lcol = (tid & 15) * 8;    // element offset within stage
    const __bf16* aload = A + (size_t)(m0 + lrow) * K + lcol;

    const __bf16* bptr = Bm + (size_t)(n0 + row) * K + half * 16;

    v8f acc0 = {}, acc1 = {}, acc2 = {}, acc3 = {};

    const int nstages = K / STAGE_COLS;
    copy16_g2l(aload, &atile[0][lrow * LDS_ROW + lcol]);

    for (int s = 0; s < nstages; ++s) {
        wait_async_lds();
        __syncthreads();
        if (s + 1 < nstages)
            copy16_g2l(aload + (size_t)(s + 1) * STAGE_COLS,
                       &atile[(s + 1) & 1][lrow * LDS_ROW + lcol]);

        const __bf16* lbase = &atile[s & 1][row * LDS_ROW + half * 16];
        const __bf16* bbase = bptr + (size_t)s * STAGE_COLS;

        // register-pipelined inner loop: 4 k-steps of 32
        v16bf af[2], bf[2][4];
        af[0]    = *(const v16bf*)(lbase);
        bf[0][0] = ldfrag(bbase);
        bf[0][1] = ldfrag(bbase + (size_t)16 * K);
        bf[0][2] = ldfrag(bbase + (size_t)32 * K);
        bf[0][3] = ldfrag(bbase + (size_t)48 * K);
        #pragma unroll
        for (int k = 0; k < 4; ++k) {
            const int cur = k & 1, nxt = cur ^ 1;
            if (k < 3) {
                const __bf16* bk = bbase + (k + 1) * 32;
                af[nxt]    = *(const v16bf*)(lbase + (k + 1) * 32);
                bf[nxt][0] = ldfrag(bk);
                bf[nxt][1] = ldfrag(bk + (size_t)16 * K);
                bf[nxt][2] = ldfrag(bk + (size_t)32 * K);
                bf[nxt][3] = ldfrag(bk + (size_t)48 * K);
            }
            acc0 = WMMA_BF16(af[cur], bf[cur][0], acc0);
            acc1 = WMMA_BF16(af[cur], bf[cur][1], acc1);
            acc2 = WMMA_BF16(af[cur], bf[cur][2], acc2);
            acc3 = WMMA_BF16(af[cur], bf[cur][3], acc3);
        }
        __syncthreads();
    }

    #pragma unroll
    for (int r = 0; r < 8; ++r) {
        int m = m0 + r + half * 8;
        float* crow = C + (size_t)m * N + n0 + row;
        crow[0]  = acc0[r];
        crow[16] = acc1[r];
        crow[32] = acc2[r];
        crow[48] = acc3[r];
    }
}

// ---------------------------------------------------------------------------
// RoPE + pack f32 -> bf16.  src/dst: [M_TOT, ncols] row-major, ncols = nheads*128.
// ---------------------------------------------------------------------------
__global__ void rope_pack_kernel(const float* __restrict__ src, __bf16* __restrict__ dst,
                                 int rows, int ncols) {
    int nheads = ncols / HD;
    int total  = rows * nheads * 64;
    int stride = gridDim.x * blockDim.x;
    for (int i = blockIdx.x * blockDim.x + threadIdx.x; i < total; i += stride) {
        int d2 = i & 63;
        int t  = i >> 6;
        int h  = t % nheads;
        int r  = t / nheads;          // r = b*SEQ + s
        int s  = r & (SEQ - 1);

        const float* p = src + (size_t)r * ncols + h * HD;
        float x1 = p[d2];
        float x2 = p[d2 + 64];
        float freq = __powf(10000.f, -(float)d2 * (1.f / 64.f));
        float ang  = (float)s * freq;
        float sn, cs;
        __sincosf(ang, &sn, &cs);

        __bf16* q = dst + (size_t)r * ncols + h * HD;
        q[d2]      = (__bf16)(x1 * cs - x2 * sn);
        q[d2 + 64] = (__bf16)(x2 * cs + x1 * sn);
    }
}

// ---------------------------------------------------------------------------
// V pack + transpose: Vf [B*S, KVD] f32 -> Vt [B, NKV, HD, SEQ] bf16
// ---------------------------------------------------------------------------
__global__ void v_pack_kernel(const float* __restrict__ Vf, __bf16* __restrict__ Vt) {
    int total  = M_TOT * KVD;
    int stride = gridDim.x * blockDim.x;
    for (int i = blockIdx.x * blockDim.x + threadIdx.x; i < total; i += stride) {
        int s   = i & (SEQ - 1);      // fastest: coalesced writes
        int t   = i >> 11;            // /SEQ
        int d   = t % HD;
        int t2  = t / HD;
        int kv  = t2 % NKV;
        int b   = t2 / NKV;
        float v = Vf[((size_t)(b * SEQ + s)) * KVD + kv * HD + d];
        Vt[i] = (__bf16)v;
    }
}

// ---------------------------------------------------------------------------
// Causal flash attention with GQA.  One wave per 16-query tile.
//   Qb:  [B*S, DIMX]  bf16 (post-RoPE)
//   Kb:  [B*S, KVD]   bf16 (post-RoPE)
//   Vt:  [B, NKV, HD, SEQ] bf16
//   ctx: [B*S, DIMX]  bf16
// ---------------------------------------------------------------------------
__global__ void __launch_bounds__(32)
flash_attn_kernel(const __bf16* __restrict__ Qb, const __bf16* __restrict__ Kb,
                  const __bf16* __restrict__ Vt, __bf16* __restrict__ ctx) {
    __shared__ __align__(32) __bf16 plds[16 * 32];   // 16 q-rows x 32 keys

    const int bid  = blockIdx.x;
    const int qt   = bid & 127;              // SEQ/16 = 128 tiles
    const int h    = (bid >> 7) & 15;
    const int b    = bid >> 11;
    const int kv   = h >> 2;                 // GQA: h / N_REP
    const int lane = threadIdx.x & 31;
    const int row  = lane & 15;
    const int half = lane >> 4;
    const int qbase = qt * 16;

    // Q fragments: 4 chunks of 32 along hd
    const __bf16* qptr = Qb + ((size_t)(b * SEQ + qbase + row)) * DIMX + h * HD + half * 16;
    v16bf qf[4];
    #pragma unroll
    for (int c = 0; c < 4; ++c) qf[c] = ldfrag(qptr + c * 32);

    v8f o[8];
    #pragma unroll
    for (int c = 0; c < 8; ++c) o[c] = (v8f){};
    float mrow[8], lrow[8];
    #pragma unroll
    for (int r = 0; r < 8; ++r) { mrow[r] = -1e30f; lrow[r] = 0.f; }

    const float scale = 0.08838834764831845f;   // 1/sqrt(128)
    const int nkt = (qbase + 15) / 32 + 1;      // causal key-tile count (32 keys)

    for (int kt = 0; kt < nkt; ++kt) {
        const int kbase = kt * 32;

        // ---- preload all K fragments, then S = Q K^T ----------------------
        const __bf16* kp0 = Kb + ((size_t)(b * SEQ + kbase + row)) * KVD + kv * HD + half * 16;
        const __bf16* kp1 = kp0 + (size_t)16 * KVD;
        v16bf kf0[4], kf1[4];
        #pragma unroll
        for (int c = 0; c < 4; ++c) kf0[c] = ldfrag(kp0 + c * 32);
        #pragma unroll
        for (int c = 0; c < 4; ++c) kf1[c] = ldfrag(kp1 + c * 32);

        v8f s0 = {}, s1 = {};
        #pragma unroll
        for (int c = 0; c < 4; ++c) s0 = WMMA_BF16(qf[c], kf0[c], s0);
        #pragma unroll
        for (int c = 0; c < 4; ++c) s1 = WMMA_BF16(qf[c], kf1[c], s1);

        // ---- issue V loads early: latency hides under softmax VALU --------
        const __bf16* vrow = Vt + (((size_t)(b * NKV + kv) * HD + row)) * SEQ + kbase + half * 16;
        v16bf vf[8];
        #pragma unroll
        for (int c = 0; c < 8; ++c) vf[c] = ldfrag(vrow + (size_t)(c * 16) * SEQ);

        // ---- online softmax (rows per-VGPR across 16-lane half-groups) ----
        const bool need_mask = (kbase + 31 > qbase);
        #pragma unroll
        for (int r = 0; r < 8; ++r) {
            int q  = qbase + r + half * 8;
            float v0 = s0[r] * scale;
            float v1 = s1[r] * scale;
            if (need_mask) {
                if (kbase + row > q)      v0 = -1e30f;
                if (kbase + 16 + row > q) v1 = -1e30f;
            }
            float mloc = fmaxf(v0, v1);
            #pragma unroll
            for (int mk = 1; mk < 16; mk <<= 1) mloc = fmaxf(mloc, __shfl_xor(mloc, mk));
            float mnew  = fmaxf(mrow[r], mloc);
            float alpha = __expf(mrow[r] - mnew);
            mrow[r] = mnew;

            float p0 = __expf(v0 - mnew);
            float p1 = __expf(v1 - mnew);
            float ls = p0 + p1;
            #pragma unroll
            for (int mk = 1; mk < 16; mk <<= 1) ls += __shfl_xor(ls, mk);
            lrow[r] = lrow[r] * alpha + ls;

            #pragma unroll
            for (int c = 0; c < 8; ++c) o[c][r] *= alpha;

            int prow = r + half * 8;
            plds[prow * 32 + row]      = (__bf16)p0;
            plds[prow * 32 + 16 + row] = (__bf16)p1;
        }

        // ---- P·V: P (16x32) from LDS as A-fragment ------------------------
        v16bf pf = *(const v16bf*)&plds[row * 32 + half * 16];
        #pragma unroll
        for (int c = 0; c < 8; ++c) o[c] = WMMA_BF16(pf, vf[c], o[c]);
    }

    // ---- normalize and store context (bf16) -------------------------------
    #pragma unroll
    for (int r = 0; r < 8; ++r) {
        float inv = 1.f / lrow[r];
        int m = qbase + r + half * 8;
        __bf16* out = ctx + ((size_t)(b * SEQ + m)) * DIMX + h * HD + row;
        #pragma unroll
        for (int c = 0; c < 8; ++c) out[c * 16] = (__bf16)(o[c][r] * inv);
    }
}

// ---------------------------------------------------------------------------
// Host-side launcher
// ---------------------------------------------------------------------------
static inline int imin(int a, int b) { return a < b ? a : b; }

extern "C" void kernel_launch(void* const* d_in, const int* in_sizes, int n_in,
                              void* d_out, int out_size, void* d_ws, size_t ws_size,
                              hipStream_t stream) {
    (void)in_sizes; (void)n_in; (void)out_size; (void)ws_size;

    const float* x  = (const float*)d_in[0];
    const float* Wq = (const float*)d_in[1];
    const float* Wk = (const float*)d_in[2];
    const float* Wv = (const float*)d_in[3];
    const float* Wo = (const float*)d_in[4];
    float* out = (float*)d_out;

    char* w = (char*)d_ws;
    auto take = [&](size_t bytes) { void* p = (void*)w; w += bytes; return p; };

    __bf16* xb  = (__bf16*)take((size_t)M_TOT * DIMX * 2);
    __bf16* wqb = (__bf16*)take((size_t)DIMX * DIMX * 2);
    __bf16* wkb = (__bf16*)take((size_t)KVD * DIMX * 2);
    __bf16* wvb = (__bf16*)take((size_t)KVD * DIMX * 2);
    __bf16* wob = (__bf16*)take((size_t)DIMX * DIMX * 2);
    __bf16* qb  = (__bf16*)take((size_t)M_TOT * DIMX * 2);
    __bf16* kb  = (__bf16*)take((size_t)M_TOT * KVD * 2);
    __bf16* vt  = (__bf16*)take((size_t)M_TOT * KVD * 2);
    __bf16* ctx = (__bf16*)take((size_t)M_TOT * DIMX * 2);
    float*  tmpf = (float*)take((size_t)M_TOT * DIMX * 4);  // reused f32 staging

    auto cvt = [&](const float* s, __bf16* d, size_t n) {
        int n4 = (int)(n / 4);
        int blocks = imin((n4 + 255) / 256, 8192);
        cvt_bf16_kernel<<<blocks, 256, 0, stream>>>(s, d, n4);
    };

    // 1) convert inputs to bf16
    cvt(x,  xb,  (size_t)M_TOT * DIMX);
    cvt(Wq, wqb, (size_t)DIMX * DIMX);
    cvt(Wk, wkb, (size_t)KVD * DIMX);
    cvt(Wv, wvb, (size_t)KVD * DIMX);
    cvt(Wo, wob, (size_t)DIMX * DIMX);

    // 2) Q projection -> RoPE pack
    gemm_bt_bf16<<<dim3(DIMX / 512, M_TOT / 16), 256, 0, stream>>>(xb, wqb, tmpf, M_TOT, DIMX, DIMX);
    rope_pack_kernel<<<4096, 256, 0, stream>>>(tmpf, qb, M_TOT, DIMX);

    // 3) K projection -> RoPE pack
    gemm_bt_bf16<<<dim3(1, M_TOT / 16), 256, 0, stream>>>(xb, wkb, tmpf, M_TOT, KVD, DIMX);
    rope_pack_kernel<<<1024, 256, 0, stream>>>(tmpf, kb, M_TOT, KVD);

    // 4) V projection -> transpose pack
    gemm_bt_bf16<<<dim3(1, M_TOT / 16), 256, 0, stream>>>(xb, wvb, tmpf, M_TOT, KVD, DIMX);
    v_pack_kernel<<<2048, 256, 0, stream>>>(tmpf, vt);

    // 5) causal flash attention (one wave per 16-query tile)
    flash_attn_kernel<<<BSZ * NH * (SEQ / 16), 32, 0, stream>>>(qb, kb, vt, ctx);

    // 6) output projection -> f32 result
    gemm_bt_bf16<<<dim3(DIMX / 512, M_TOT / 16), 256, 0, stream>>>(ctx, wob, out, M_TOT, DIMX, DIMX);
}
